// LEFTNet_73641509257758
// MI455X (gfx1250) — matrix operand order repaired and checked by reading, not compiled, not run on hardware
//
#include <hip/hip_runtime.h>

// ---------------- problem constants (from reference) ----------------
#define NN 10000
#define EE 320000
#define HH 128
#define RR 32
#define H3 384          // 3*H
#define KW 416          // 3*H + R
#define INV_SQRT_3 0.57735026918962576f
#define INV_SQRT_H 0.08838834764831845f   // 1/sqrt(128)

// ---------------- WMMA fragment plumbing (CDNA5 wave32) ----------------
typedef __attribute__((ext_vector_type(16))) __bf16 v16bf;
typedef __attribute__((ext_vector_type(8)))  float  v8f;
typedef __attribute__((ext_vector_type(4)))  unsigned v4u;
typedef __attribute__((ext_vector_type(8)))  int      v8i;
typedef __attribute__((ext_vector_type(4)))  int      v4i;

union Frag16 { v16bf v; unsigned u[8]; };
union FragC  { v8f  v;  float    f[8]; };

__device__ __forceinline__ unsigned short f2bf(float f) {
  unsigned x = __float_as_uint(f);
  return (unsigned short)((x + 0x7FFFu + ((x >> 16) & 1u)) >> 16); // RNE-ish
}
__device__ __forceinline__ unsigned pack2(float a, float b) {
  return (unsigned)f2bf(a) | ((unsigned)f2bf(b) << 16);
}

// padded LDS strides (ushort units): 80B rows = 20 dwords -> conflict-free frag
// loads, and 16B-aligned rows for b128 copies.
#define AS_STRIDE 40
#define BS_STRIDE 40
#define NH_STRIDE 132    // node hidden (128 cols)
#define EH_STRIDE 388    // edge hidden (384 cols)

// A-matrix 16x32 bf16 frag from row-major LDS tile.
// ISA layout: lanes 0-15 hold K {0..7,16..23}; lanes 16-31 hold K {8..15,24..31}.
__device__ __forceinline__ void load_a(Frag16& fa, const unsigned short* lds,
                                       int rowBase, int stride) {
  const int lane = threadIdx.x & 31;
  const int hi = lane >> 4;
  const unsigned short* p = lds + (rowBase + (lane & 15)) * stride;
#pragma unroll
  for (int jj = 0; jj < 8; ++jj) {
    const int j = 2 * jj;
    const int k = (j < 8) ? (8 * hi + j) : (8 + 8 * hi + j);
    fa.u[jj] = *(const unsigned*)(p + k);
  }
}

// B-matrix 32x16 bf16 frag from *transposed* LDS tile ([N][K] layout).
// ISA layout: lanes 0-15 hold K {0..7,16..23}; lanes 16-31 hold K {8..15,24..31}.
__device__ __forceinline__ void load_b(Frag16& fb, const unsigned short* ldsT,
                                       int colBase) {
  const int lane = threadIdx.x & 31;
  const int hi = lane >> 4;
  const unsigned short* p = ldsT + (colBase + (lane & 15)) * BS_STRIDE + 8 * hi;
#pragma unroll
  for (int jj = 0; jj < 8; ++jj)
    fb.u[jj] = *(const unsigned*)(p + 2 * jj + (jj >= 4 ? 8 : 0));
}

__device__ __forceinline__ v8f wmma_bf16(const Frag16& a, const Frag16& b, v8f c) {
  return __builtin_amdgcn_wmma_f32_16x16x32_bf16(false, a.v, false, b.v,
                                                 (short)0, c, false, false);
}

__device__ __forceinline__ float silu_f(float v) {
  return v * (1.0f / (1.0f + __expf(-v)));
}

__device__ __forceinline__ unsigned rfl(unsigned x) {
  return (unsigned)__builtin_amdgcn_readfirstlane((int)x);
}

// ---- Tensor Data Mover: DMA one [nrows x 32] bf16 tile (row stride ktot
// elements in global, padded 80B rows in LDS) to lds_off. Issue only (no wait).
// D# per CDNA5 ISA ch.8: group0 = {flags, lds_addr, global_addr, type=2},
// group1 = {data_size=2B, pad 16dw->+4dw, dims/strides}.
#if defined(__has_builtin)
#if __has_builtin(__builtin_amdgcn_tensor_load_to_lds)
#define HAVE_TDM 1
#endif
#endif

__device__ __forceinline__ void tdm_issue(unsigned lds_off,
                                          const unsigned short* gsrc,
                                          int ktot, int nrows) {
#ifdef HAVE_TDM
  const unsigned long long ga = (unsigned long long)(uintptr_t)gsrc;
  v4u g0 = { rfl(1u),                                   // count=1, user mode
             rfl(lds_off),                              // lds_addr
             rfl((unsigned)(ga & 0xFFFFFFFFu)),         // global_addr[31:0]
             rfl(((unsigned)(ga >> 32) & 0x01FFFFFFu) | (2u << 30)) }; // +type=2
  const unsigned w0 = (1u << 16)            // data_size = 2 bytes
                    | (1u << 20)            // pad_enable
                    | (3u << 22)            // pad_interval: 16 dwords (64B row)
                    | (3u << 25);           // pad_amount: 4 dwords (+16B -> 80B)
  v8i g1 = { (int)rfl(w0),
             (int)rfl(((unsigned)ktot & 0xFFFFu) << 16),            // dim0 lo16
             (int)rfl((((unsigned)ktot >> 16) & 0xFFFFu) |
                      (((unsigned)nrows & 0xFFFFu) << 16)),         // dim0 hi, dim1 lo
             (int)rfl((((unsigned)nrows >> 16) & 0xFFFFu) |
                      (32u << 16)),                                  // dim1 hi, tile_dim0=32
             (int)rfl((unsigned)nrows),                              // tile_dim1, tile_dim2=0
             (int)rfl((unsigned)ktot),                               // dim0_stride lo32
             0, 0 };
  v4i z4 = { 0, 0, 0, 0 };
#if defined(__clang_major__) && (__clang_major__ >= 23)
  v8i z8 = { 0, 0, 0, 0, 0, 0, 0, 0 };
  __builtin_amdgcn_tensor_load_to_lds(g0, g1, z4, z4, z8, 0);
#else
  __builtin_amdgcn_tensor_load_to_lds(g0, g1, z4, z4, 0);
#endif
#endif
}

// Simple (non-pipelined) staging used by the node kernel / TDM fallback.
__device__ __forceinline__ void stage_b_chunk(unsigned short* Bs,
                                              const unsigned short* gsrc,
                                              int ktot, int nrows,
                                              int tid, int wave) {
#ifdef HAVE_TDM
  if (wave == 0) {
    tdm_issue((unsigned)(uintptr_t)(void*)Bs, gsrc, ktot, nrows);
    __builtin_amdgcn_s_wait_tensorcnt(0);
  }
#else
  for (int i = tid; i < nrows * 2; i += 256) {   // 2 x uint4 (32B) per 64B row
    const int n = i >> 1, q = i & 1;
    *(uint4*)&Bs[n * BS_STRIDE + q * 16] =
        *(const uint4*)&gsrc[(size_t)n * ktot + q * 16];
  }
#endif
}

// ---------------- kernel 0: zero the output ----------------
__global__ void zero_kernel(float* __restrict__ p, int n) {
  int i = blockIdx.x * 256 + threadIdx.x;
  if (i < n) p[i] = 0.0f;
}

// ---------------- kernel 0b: convert+transpose f32 [K][N] -> bf16 [N][K] ----------------
__global__ void convT_kernel(const float* __restrict__ in,
                             unsigned short* __restrict__ out, int K, int Nc) {
  int i = blockIdx.x * 256 + threadIdx.x;
  if (i < K * Nc) {
    const int k = i / Nc, n = i % Nc;
    out[(size_t)n * K + k] = f2bf(in[i]);
  }
}

// ---------------- kernel 1: xh = silu(x@Wx1+bx1)@Wx2+bx2 -> d_ws ----------------
__global__ __launch_bounds__(256) void node_xh_kernel(
    const float* __restrict__ x,
    const unsigned short* __restrict__ Wx1T, const float* __restrict__ bx1,
    const unsigned short* __restrict__ Wx2T, const float* __restrict__ bx2,
    float* __restrict__ xh) {
  __shared__ __align__(16) unsigned char blob[2560 + 30720 + 8448];
  unsigned short* As = (unsigned short*)blob;                   // 32 x 32 (pad 40)
  unsigned short* Bs = (unsigned short*)(blob + 2560);          // <=384 x 32 (pad 40), [N][K]
  unsigned short* Hs = (unsigned short*)(blob + 2560 + 30720);  // 32 x 128 (pad 132)

  const int tid = threadIdx.x;
  const int lane = tid & 31;
  const int wave = tid >> 5;
  const int wm = wave >> 2;   // 0..1 : 16-row group
  const int wn = wave & 3;    // 0..3 : column group
  const int hi = lane >> 4;
  const int r0 = blockIdx.x * 32;

  // ---- stage 1: hidden = silu(x @ Wx1 + bx1), K = 128, N = 128 ----
  FragC acc1[2];
#pragma unroll
  for (int t = 0; t < 2; ++t)
#pragma unroll
    for (int g = 0; g < 8; ++g) acc1[t].f[g] = 0.0f;

#pragma unroll 1
  for (int kc = 0; kc < 4; ++kc) {
    const int k0 = kc * 32;
    stage_b_chunk(Bs, Wx1T + k0, HH, HH, tid, wave);
    for (int i = tid; i < 32 * 8; i += 256) {
      const int r = i >> 3, cq = (i & 7) * 4;
      const int row = r0 + r;
      float4 v = make_float4(0.f, 0.f, 0.f, 0.f);
      if (row < NN) v = *(const float4*)&x[(size_t)row * HH + k0 + cq];
      *(uint2*)&As[r * AS_STRIDE + cq] = make_uint2(pack2(v.x, v.y), pack2(v.z, v.w));
    }
    __syncthreads();
    Frag16 fa; load_a(fa, As, wm * 16, AS_STRIDE);
#pragma unroll
    for (int t = 0; t < 2; ++t) {
      Frag16 fb; load_b(fb, Bs, wn * 32 + t * 16);
      acc1[t].v = wmma_bf16(fa, fb, acc1[t].v);
    }
    __syncthreads();
  }
#pragma unroll
  for (int t = 0; t < 2; ++t) {
    const int n = wn * 32 + t * 16 + (lane & 15);
    const float b = bx1[n];
#pragma unroll
    for (int g = 0; g < 8; ++g) {
      const int R = wm * 16 + g + 8 * hi;
      Hs[R * NH_STRIDE + n] = f2bf(silu_f(acc1[t].f[g] + b));
    }
  }
  __syncthreads();

  // ---- stage 2: xh = hidden @ Wx2 + bx2, K = 128, N = 384 ----
  FragC acc2[6];
#pragma unroll
  for (int t = 0; t < 6; ++t)
#pragma unroll
    for (int g = 0; g < 8; ++g) acc2[t].f[g] = 0.0f;

#pragma unroll 1
  for (int kc = 0; kc < 4; ++kc) {
    const int k0 = kc * 32;
    stage_b_chunk(Bs, Wx2T + k0, HH, H3, tid, wave);
    __syncthreads();
    Frag16 fa; load_a(fa, Hs + k0, wm * 16, NH_STRIDE);
#pragma unroll
    for (int t = 0; t < 6; ++t) {
      Frag16 fb; load_b(fb, Bs, wn * 96 + t * 16);
      acc2[t].v = wmma_bf16(fa, fb, acc2[t].v);
    }
    __syncthreads();
  }
#pragma unroll
  for (int t = 0; t < 6; ++t) {
    const int n = wn * 96 + t * 16 + (lane & 15);
    const float b = bx2[n];
#pragma unroll
    for (int g = 0; g < 8; ++g) {
      const int R = wm * 16 + g + 8 * hi;
      const int row = r0 + R;
      if (row < NN) xh[(size_t)row * H3 + n] = acc2[t].f[g] + b;
    }
  }
}

// ---------------- kernel 2: fused edge pipeline + scatter ----------------
// B-chunk schedule: i=0 -> WrT (ktot=32); i=1..13 -> Wi1T (ktot=416);
// i=14..25 -> Wi2T (ktot=384). All tiles are [384 x 32] bf16.
#define NCHUNK 26

__device__ __forceinline__ void issue_edge_chunk(int i, unsigned bs0, unsigned bs1,
                                                 const unsigned short* WrT,
                                                 const unsigned short* Wi1T,
                                                 const unsigned short* Wi2T) {
  const unsigned off = (i & 1) ? bs1 : bs0;
  if (i == 0)       tdm_issue(off, WrT, RR, H3);
  else if (i <= 13) tdm_issue(off, Wi1T + (i - 1) * 32, KW, H3);
  else              tdm_issue(off, Wi2T + (i - 14) * 32, H3, H3);
}

// Depth-2 pipeline advance: issue chunk i+1, then wait until chunk i has
// landed (TDM ops of one wave complete in order -> TENSORcnt<=1 suffices).
__device__ __forceinline__ void pipe_advance(int i, unsigned bs0, unsigned bs1,
                                             const unsigned short* WrT,
                                             const unsigned short* Wi1T,
                                             const unsigned short* Wi2T, int wave) {
#ifdef HAVE_TDM
  if (wave == 0) {
    if (i + 1 < NCHUNK) {
      issue_edge_chunk(i + 1, bs0, bs1, WrT, Wi1T, Wi2T);
      __builtin_amdgcn_s_wait_tensorcnt(1);
    } else {
      __builtin_amdgcn_s_wait_tensorcnt(0);
    }
  }
#endif
}

#ifndef HAVE_TDM
__device__ __forceinline__ void manual_b_copy(unsigned short* Bs,
                                              const unsigned short* gsrc,
                                              int ktot, int tid) {
  for (int i = tid; i < H3 * 2; i += 256) {
    const int n = i >> 1, q = i & 1;
    *(uint4*)&Bs[n * BS_STRIDE + q * 16] =
        *(const uint4*)&gsrc[(size_t)n * ktot + q * 16];
  }
}
#endif

__global__ __launch_bounds__(256) void edge_kernel(
    const float* __restrict__ edge_rbf, const float* __restrict__ weight,
    const float* __restrict__ edge_vector,
    const unsigned short* __restrict__ WrT,  const float* __restrict__ br,
    const unsigned short* __restrict__ Wi1T, const float* __restrict__ bi1,
    const unsigned short* __restrict__ Wi2T, const float* __restrict__ bi2,
    const int*   __restrict__ edge_index,
    const float* __restrict__ xh,  const float* __restrict__ vec,
    float* __restrict__ dx, float* __restrict__ dvec) {
  // As @0 (2560) | Bs0 @2560 (30720) | Bs1 @33280 (30720) | Hs @64000 (24832)
  __shared__ __align__(16) unsigned char blob[2560 + 2 * 30720 + 24832]; // 88832 B
  unsigned short* As  = (unsigned short*)blob;
  unsigned short* Bs0 = (unsigned short*)(blob + 2560);
  unsigned short* Bs1 = (unsigned short*)(blob + 33280);
  unsigned short* Hs  = (unsigned short*)(blob + 64000);
  float*          Ms  = (float*)blob;            // 32 x 384 f32 (reuses front)

  const int tid = threadIdx.x;
  const int lane = tid & 31;
  const int wave = tid >> 5;
  const int wm = wave >> 2;
  const int wn = wave & 3;
  const int hi = lane >> 4;
  const int e0 = blockIdx.x * 32;   // E is an exact multiple of 32

  const unsigned bs0 = (unsigned)(uintptr_t)(void*)Bs0;
  const unsigned bs1 = (unsigned)(uintptr_t)(void*)Bs1;

#ifdef HAVE_TDM
  if (wave == 0) issue_edge_chunk(0, bs0, bs1, WrT, Wi1T, Wi2T);  // prologue
#endif

  // ---- stage 0 (chunk 0): rbf_acc = edge_rbf @ Wr ----
  FragC rbf[6];
#pragma unroll
  for (int t = 0; t < 6; ++t)
#pragma unroll
    for (int g = 0; g < 8; ++g) rbf[t].f[g] = 0.0f;

  for (int i = tid; i < 32 * 8; i += 256) {
    const int r = i >> 3, cq = (i & 7) * 4;
    const float4 v = *(const float4*)&edge_rbf[(size_t)(e0 + r) * RR + cq];
    *(uint2*)&As[r * AS_STRIDE + cq] = make_uint2(pack2(v.x, v.y), pack2(v.z, v.w));
  }
#ifndef HAVE_TDM
  manual_b_copy(Bs0, WrT, RR, tid);
#endif
  pipe_advance(0, bs0, bs1, WrT, Wi1T, Wi2T, wave);
  __syncthreads();
  {
    Frag16 fa; load_a(fa, As, wm * 16, AS_STRIDE);
#pragma unroll
    for (int t = 0; t < 6; ++t) {
      Frag16 fb; load_b(fb, Bs0, wn * 96 + t * 16);
      rbf[t].v = wmma_bf16(fa, fb, rbf[t].v);
    }
  }
  __syncthreads();

  // ---- stage 1 (chunks 1..13): hidden = silu(weight @ Wi1 + bi1) ----
  FragC acc[6];
#pragma unroll
  for (int t = 0; t < 6; ++t)
#pragma unroll
    for (int g = 0; g < 8; ++g) acc[t].f[g] = 0.0f;

#pragma unroll 1
  for (int kc = 0; kc < 13; ++kc) {
    const int ci = 1 + kc;
    const int k0 = kc * 32;
    for (int i = tid; i < 32 * 8; i += 256) {
      const int r = i >> 3, cq = (i & 7) * 4;
      const float4 v = *(const float4*)&weight[(size_t)(e0 + r) * KW + k0 + cq];
      *(uint2*)&As[r * AS_STRIDE + cq] = make_uint2(pack2(v.x, v.y), pack2(v.z, v.w));
    }
    if (kc + 1 < 13)  // lowers to global_prefetch_b8
      __builtin_prefetch((const void*)&weight[(size_t)(e0 + (tid >> 3)) * KW + k0 + 32], 0, 1);
#ifndef HAVE_TDM
    manual_b_copy(Bs0, Wi1T + k0, KW, tid);
#endif
    pipe_advance(ci, bs0, bs1, WrT, Wi1T, Wi2T, wave);
    __syncthreads();
    const unsigned short* Bc = (ci & 1) ? Bs1 : Bs0;
#ifndef HAVE_TDM
    Bc = Bs0;
#endif
    Frag16 fa; load_a(fa, As, wm * 16, AS_STRIDE);
#pragma unroll
    for (int t = 0; t < 6; ++t) {
      Frag16 fb; load_b(fb, Bc, wn * 96 + t * 16);
      acc[t].v = wmma_bf16(fa, fb, acc[t].v);
    }
    __syncthreads();
  }
  // bias + SiLU -> bf16 hidden tile in LDS (chunk 14's DMA already in flight)
#pragma unroll
  for (int t = 0; t < 6; ++t) {
    const int n = wn * 96 + t * 16 + (lane & 15);
    const float b = bi1[n];
#pragma unroll
    for (int g = 0; g < 8; ++g) {
      const int R = wm * 16 + g + 8 * hi;
      Hs[R * EH_STRIDE + n] = f2bf(silu_f(acc[t].f[g] + b));
    }
  }
  __syncthreads();

  // ---- stage 2 (chunks 14..25): w = hidden @ Wi2 + bi2 ----
#pragma unroll
  for (int t = 0; t < 6; ++t)
#pragma unroll
    for (int g = 0; g < 8; ++g) acc[t].f[g] = 0.0f;

#pragma unroll 1
  for (int kc = 0; kc < 12; ++kc) {
    const int ci = 14 + kc;
    const int k0 = kc * 32;
#ifndef HAVE_TDM
    manual_b_copy(Bs0, Wi2T + k0, H3, tid);
#endif
    pipe_advance(ci, bs0, bs1, WrT, Wi1T, Wi2T, wave);
    __syncthreads();
    const unsigned short* Bc = (ci & 1) ? Bs1 : Bs0;
#ifndef HAVE_TDM
    Bc = Bs0;
#endif
    Frag16 fa; load_a(fa, Hs + k0, wm * 16, EH_STRIDE);
#pragma unroll
    for (int t = 0; t < 6; ++t) {
      Frag16 fb; load_b(fb, Bc, wn * 96 + t * 16);
      acc[t].v = wmma_bf16(fa, fb, acc[t].v);
    }
    __syncthreads();
  }

  // ---- combine: m_pre = (rbf + br) * (w + bi2) -> f32 LDS (reuses blob) ----
#pragma unroll
  for (int t = 0; t < 6; ++t) {
    const int n = wn * 96 + t * 16 + (lane & 15);
    const float brv = br[n];
    const float b2v = bi2[n];
#pragma unroll
    for (int g = 0; g < 8; ++g) {
      const int R = wm * 16 + g + 8 * hi;
      Ms[R * H3 + n] = (rbf[t].f[g] + brv) * (acc[t].f[g] + b2v);
    }
  }
  __syncthreads();

  // ---- epilogue: gather xh[src], split, scale, scatter-add ----
  const int* __restrict__ srcp = edge_index;
  const int* __restrict__ dstp = edge_index + EE;
  for (int i = tid; i < 32 * HH; i += 256) {
    const int r = i >> 7;          // edge row within block
    const int h = i & 127;         // channel
    const int e = e0 + r;
    const int s = srcp[e];
    const int d = dstp[e];
    const float* xs = xh + (size_t)s * H3;
    const float x1  = Ms[r * H3 + h]            * xs[h];
    const float m2  = Ms[r * H3 + HH + h]       * xs[HH + h] * INV_SQRT_3;
    const float m3  = Ms[r * H3 + 2 * HH + h]   * xs[2 * HH + h];
    atomicAdd(&dx[(size_t)d * HH + h], x1);
    const float ev0 = edge_vector[e * 3 + 0];
    const float ev1 = edge_vector[e * 3 + 1];
    const float ev2 = edge_vector[e * 3 + 2];
    const float* vs = vec + (size_t)s * H3;
    atomicAdd(&dvec[(size_t)d * H3 + 0 * HH + h], (vs[0 * HH + h] * m2 + m3 * ev0) * INV_SQRT_H);
    atomicAdd(&dvec[(size_t)d * H3 + 1 * HH + h], (vs[1 * HH + h] * m2 + m3 * ev1) * INV_SQRT_H);
    atomicAdd(&dvec[(size_t)d * H3 + 2 * HH + h], (vs[2 * HH + h] * m2 + m3 * ev2) * INV_SQRT_H);
  }
}

// ---------------- launcher ----------------
extern "C" void kernel_launch(void* const* d_in, const int* in_sizes, int n_in,
                              void* d_out, int out_size, void* d_ws, size_t ws_size,
                              hipStream_t stream) {
  const float* x           = (const float*)d_in[0];
  const float* vec         = (const float*)d_in[1];
  const float* edge_rbf    = (const float*)d_in[2];
  const float* weight      = (const float*)d_in[3];
  const float* edge_vector = (const float*)d_in[4];
  const float* Wx1         = (const float*)d_in[5];
  const float* bx1         = (const float*)d_in[6];
  const float* Wx2         = (const float*)d_in[7];
  const float* bx2         = (const float*)d_in[8];
  const float* Wr          = (const float*)d_in[9];
  const float* br          = (const float*)d_in[10];
  const float* Wi1         = (const float*)d_in[11];
  const float* bi1         = (const float*)d_in[12];
  const float* Wi2         = (const float*)d_in[13];
  const float* bi2         = (const float*)d_in[14];
  const int*   edge_index  = (const int*)d_in[15];

  // workspace layout: xh f32 [N,3H] then bf16 transposed weights
  float* xh = (float*)d_ws;
  unsigned short* bfb =
      (unsigned short*)((char*)d_ws + (((size_t)NN * H3 * 4 + 255) & ~(size_t)255));
  unsigned short* WrT  = bfb;                       // [384][32]
  unsigned short* Wi1T = WrT + (size_t)H3 * RR;     // [384][416]
  unsigned short* Wi2T = Wi1T + (size_t)H3 * KW;    // [384][384]
  unsigned short* Wx1T = Wi2T + (size_t)H3 * H3;    // [128][128]
  unsigned short* Wx2T = Wx1T + (size_t)HH * HH;    // [384][128]

  float* dx   = (float*)d_out;                // [N, H]
  float* dvec = dx + (size_t)NN * HH;         // [N, 3, H]

  zero_kernel<<<(out_size + 255) / 256, 256, 0, stream>>>((float*)d_out, out_size);
  convT_kernel<<<(RR * H3 + 255) / 256, 256, 0, stream>>>(Wr,  WrT,  RR, H3);
  convT_kernel<<<(KW * H3 + 255) / 256, 256, 0, stream>>>(Wi1, Wi1T, KW, H3);
  convT_kernel<<<(H3 * H3 + 255) / 256, 256, 0, stream>>>(Wi2, Wi2T, H3, H3);
  convT_kernel<<<(HH * HH + 255) / 256, 256, 0, stream>>>(Wx1, Wx1T, HH, HH);
  convT_kernel<<<(HH * H3 + 255) / 256, 256, 0, stream>>>(Wx2, Wx2T, HH, H3);
  node_xh_kernel<<<(NN + 31) / 32, 256, 0, stream>>>(x, Wx1T, bx1, Wx2T, bx2, xh);
  edge_kernel<<<EE / 32, 256, 0, stream>>>(edge_rbf, weight, edge_vector,
                                           WrT, br, Wi1T, bi1, Wi2T, bi2,
                                           edge_index, xh, vec, dx, dvec);
}
